// Attention_49254684951026
// MI455X (gfx1250) — matrix operand order repaired and checked by reading, not compiled, not run on hardware
//
#include <hip/hip_runtime.h>
#include <hip/hip_bf16.h>

typedef __bf16 bf16;
typedef __attribute__((ext_vector_type(16))) __bf16 v16bf;
typedef __attribute__((ext_vector_type(8)))  __bf16 v8bf;
typedef __attribute__((ext_vector_type(8)))  float  v8f;

#define D_MODEL 768
#define D_HEAD  64
#define N_HEADS 12
#define BATCH   8
#define SEQ     1024
#define ROWS    (BATCH*SEQ)      /* 8192 */
#define NQKV    (3*D_MODEL)      /* 2304 */

// Full-register single-wave kernels: one wave/SIMD, no spill pressure.
#define WAVE_KERNEL __launch_bounds__(32, 1) __attribute__((amdgpu_waves_per_eu(1)))

// ---------------------------------------------------------------- WMMA helpers

__device__ __forceinline__ v8f wmma_bf16(v16bf a, v16bf b, v8f c) {
  // D = A(16x32) * B(32x16) + C, fp32 accum
  return __builtin_amdgcn_wmma_f32_16x16x32_bf16(
      /*neg_a=*/false, a, /*neg_b=*/false, b,
      /*c_mod=*/(short)0, c, /*reuse_a=*/false, /*reuse_b=*/false);
}

// A-matrix fragment (16x32 bf16, row-major src with leading dim ld).
// Lane holds row m=lane%16; lanes 0-15 K={0..15},
// lanes 16-31 K={16..31} (matches load_bT_frag).
__device__ __forceinline__ v16bf load_a_frag(const bf16* p, int ld) {
  const int lane = threadIdx.x & 31;
  const int m    = lane & 15;
  const int kg   = lane >> 4;
  const bf16* rowp = p + m * ld + 16 * kg;
  v8bf lo = *reinterpret_cast<const v8bf*>(rowp);       // K = 16*kg .. 16*kg+7
  v8bf hi = *reinterpret_cast<const v8bf*>(rowp + 8);   // K = 16*kg+8 ..
  return __builtin_shufflevector(lo, hi, 0,1,2,3,4,5,6,7,8,9,10,11,12,13,14,15);
}

// B-matrix fragment (32x16) loaded from a transposed (N-major) source:
// B(k,n) = p[n*ld + k]. Lane holds col n=lane%16, K = 16*(lane/16)+j.
// 16 contiguous bf16 per lane = one 32-byte load.
__device__ __forceinline__ v16bf load_bT_frag(const bf16* p, int ld) {
  const int lane = threadIdx.x & 31;
  const int n    = lane & 15;
  const int kg   = lane >> 4;
  return *reinterpret_cast<const v16bf*>(p + n * ld + 16 * kg);
}

// ---------------------------------------------------------------- prep kernels

__global__ void prep_x_bf16(const float* __restrict__ x, bf16* __restrict__ xb, int n) {
  for (int i = blockIdx.x * blockDim.x + threadIdx.x; i < n; i += gridDim.x * blockDim.x)
    xb[i] = (bf16)x[i];
}

// Pack W_Q,W_K,W_V [12,768,64] fp32 -> B^T layout wT[(which*768+head*64+h)][m] bf16
__global__ void pack_wqkv(const float* __restrict__ wq, const float* __restrict__ wk,
                          const float* __restrict__ wv, bf16* __restrict__ wT) {
  int i = blockIdx.x * blockDim.x + threadIdx.x;
  if (i >= NQKV * D_MODEL) return;
  int c = i / D_MODEL, m = i % D_MODEL;
  int w_sel = c / D_MODEL;                    // 0,1,2
  int rem = c % D_MODEL;
  int head = rem / D_HEAD, h = rem % D_HEAD;
  const float* w = (w_sel == 0) ? wq : (w_sel == 1) ? wk : wv;
  wT[i] = (bf16)w[((size_t)head * D_MODEL + m) * D_HEAD + h];
}

// Pack W_O [12*64, 768] fp32 -> B^T layout woT[m][nh] bf16
__global__ void pack_wo(const float* __restrict__ wo, bf16* __restrict__ wT) {
  int i = blockIdx.x * blockDim.x + threadIdx.x;
  if (i >= D_MODEL * D_MODEL) return;
  int m = i / D_MODEL, nh = i % D_MODEL;
  wT[i] = (bf16)wo[(size_t)nh * D_MODEL + m];
}

// ---------------------------------------------------------------- QKV GEMM
// One wave computes a 64x64 tile of [8192 x 2304] = x_bf16 * Wqkv.
// Per K-step: 4 A-frags + 4 B-frags (16 b128 loads) feed 16 WMMAs.
// Q,K stored [b,head,s,64]; V stored transposed [b,head,64,s].
__global__ void WAVE_KERNEL
qkv_gemm(const bf16* __restrict__ x, const bf16* __restrict__ wT,
         const float* __restrict__ bQ, const float* __restrict__ bK,
         const float* __restrict__ bV,
         bf16* __restrict__ qo, bf16* __restrict__ ko, bf16* __restrict__ vo) {
  const int row0 = blockIdx.x * 64;
  const int c0   = blockIdx.y * 64;

  v8f acc[4][4] = {};
  for (int k0 = 0; k0 < D_MODEL; k0 += 32) {
    if (k0 + 32 < D_MODEL)
      __builtin_prefetch(x + (size_t)row0 * D_MODEL + k0 + 32, 0, 3);  // WGP-scope
    v16bf a[4];
#pragma unroll
    for (int mi = 0; mi < 4; ++mi)
      a[mi] = load_a_frag(x + (size_t)(row0 + 16 * mi) * D_MODEL + k0, D_MODEL);
#pragma unroll
    for (int t = 0; t < 4; ++t) {
      v16bf b = load_bT_frag(wT + (size_t)(c0 + 16 * t) * D_MODEL + k0, D_MODEL);
#pragma unroll
      for (int mi = 0; mi < 4; ++mi)
        acc[mi][t] = wmma_bf16(a[mi], b, acc[mi][t]);
    }
  }

  const int lane = threadIdx.x & 31;
  const int n = lane & 15, g = lane >> 4;
  const int w_sel = c0 / D_MODEL;                 // 0=Q,1=K,2=V (tile never crosses)
  const int head  = (c0 % D_MODEL) / D_HEAD;
  const float* bias = (w_sel == 0) ? bQ : (w_sel == 1) ? bK : bV;

#pragma unroll
  for (int mi = 0; mi < 4; ++mi) {
#pragma unroll
    for (int t = 0; t < 4; ++t) {
      const int h = 16 * t + n;                   // 0..63 within head
      const float bv = bias[head * D_HEAD + h];
#pragma unroll
      for (int r = 0; r < 8; ++r) {
        const int row = row0 + 16 * mi + r + 8 * g;
        const int b_  = row / SEQ, s = row % SEQ;
        const float val = acc[mi][t][r] + bv;
        if (w_sel == 2) {
          vo[(((size_t)b_ * N_HEADS + head) * D_HEAD + h) * SEQ + s] = (bf16)val;
        } else {
          bf16* dst = (w_sel == 0) ? qo : ko;
          dst[(((size_t)b_ * N_HEADS + head) * SEQ + s) * D_HEAD + h] = (bf16)val;
        }
      }
    }
  }
}

// ---------------------------------------------------------------- Flash attention
// One wave per (b,head, 32 query rows). K/V fragments loaded once per 32-key
// block and reused across both 16-row query sub-tiles. Online softmax, fp32.
__global__ void WAVE_KERNEL
attn_kernel(const bf16* __restrict__ q, const bf16* __restrict__ k,
            const bf16* __restrict__ vT, bf16* __restrict__ z) {
  __shared__ __align__(32) bf16 lds_p[32 * 32];   // P tile staging (C->A relayout)

  const int q0   = blockIdx.x * 32;
  const int bh   = blockIdx.y;                    // b*12 + head
  const int head = bh % N_HEADS;
  const int b    = bh / N_HEADS;

  const bf16* qb = q  + (size_t)bh * SEQ * D_HEAD;
  const bf16* kb = k  + (size_t)bh * SEQ * D_HEAD;
  const bf16* vb = vT + (size_t)bh * D_HEAD * SEQ;

  v16bf qa[2][2];
#pragma unroll
  for (int mi = 0; mi < 2; ++mi)
#pragma unroll
    for (int kk = 0; kk < 2; ++kk)
      qa[mi][kk] = load_a_frag(qb + (size_t)(q0 + 16 * mi) * D_HEAD + 32 * kk, D_HEAD);

  const int lane = threadIdx.x & 31;
  const int n = lane & 15, g = lane >> 4;

  v8f accz[2][4] = {};
  float mst[2][8], lst[2][8];
#pragma unroll
  for (int mi = 0; mi < 2; ++mi)
#pragma unroll
    for (int r = 0; r < 8; ++r) { mst[mi][r] = -1e30f; lst[mi][r] = 0.f; }

  const int nkb = q0 / 32 + 1;                    // causal: keys <= q0+31
  for (int kb32 = 0; kb32 < nkb; ++kb32) {
    const int key0 = kb32 * 32;

    // K fragments for this key block (shared by both query sub-tiles)
    v16bf kf[2][2];
#pragma unroll
    for (int kg = 0; kg < 2; ++kg)
#pragma unroll
      for (int kk = 0; kk < 2; ++kk)
        kf[kg][kk] = load_bT_frag(kb + (size_t)(key0 + 16 * kg) * D_HEAD + 32 * kk, D_HEAD);

#pragma unroll
    for (int mi = 0; mi < 2; ++mi) {
      v8f s0 = {}, s1 = {};
      s0 = wmma_bf16(qa[mi][0], kf[0][0], s0);
      s0 = wmma_bf16(qa[mi][1], kf[0][1], s0);
      s1 = wmma_bf16(qa[mi][0], kf[1][0], s1);
      s1 = wmma_bf16(qa[mi][1], kf[1][1], s1);

      float scl[8];
#pragma unroll
      for (int r = 0; r < 8; ++r) {
        const int qi = q0 + 16 * mi + r + 8 * g;
        float v0 = s0[r] * 0.125f;                // 1/sqrt(64)
        float v1 = s1[r] * 0.125f;
        if (key0 + n      > qi) v0 = -1e30f;      // causal mask
        if (key0 + 16 + n > qi) v1 = -1e30f;

        float bm = fmaxf(v0, v1);
#pragma unroll
        for (int msk = 1; msk <= 8; msk <<= 1) bm = fmaxf(bm, __shfl_xor(bm, msk));
        const float mnew = fmaxf(mst[mi][r], bm);
        const float sc = __expf(mst[mi][r] - mnew);
        const float p0 = __expf(v0 - mnew);
        const float p1 = __expf(v1 - mnew);
        float bs = p0 + p1;
#pragma unroll
        for (int msk = 1; msk <= 8; msk <<= 1) bs += __shfl_xor(bs, msk);
        lst[mi][r] = lst[mi][r] * sc + bs;
        mst[mi][r] = mnew;
        scl[r] = sc;
        s0[r] = p0; s1[r] = p1;
      }
#pragma unroll
      for (int t = 0; t < 4; ++t)
#pragma unroll
        for (int r = 0; r < 8; ++r) accz[mi][t][r] *= scl[r];

      // P (C-layout in regs) -> LDS row-major 32x32 -> A-fragment
#pragma unroll
      for (int r = 0; r < 8; ++r) {
        lds_p[(16 * mi + r + 8 * g) * 32 + n]      = (bf16)s0[r];
        lds_p[(16 * mi + r + 8 * g) * 32 + 16 + n] = (bf16)s1[r];
      }
    }
    asm volatile("s_wait_dscnt 0" ::: "memory");  // cross-lane LDS RAW fence

    // V fragments for this key block (shared by both query sub-tiles)
    v16bf vf[4];
#pragma unroll
    for (int t = 0; t < 4; ++t)
      vf[t] = load_bT_frag(vb + (size_t)(16 * t) * SEQ + key0, SEQ);

#pragma unroll
    for (int mi = 0; mi < 2; ++mi) {
      const v16bf pa = load_a_frag(lds_p + 16 * mi * 32, 32);
#pragma unroll
      for (int t = 0; t < 4; ++t)
        accz[mi][t] = wmma_bf16(pa, vf[t], accz[mi][t]);
    }
  }

  // normalize and store z as [8192 x 768] bf16, col = head*64 + h
#pragma unroll
  for (int mi = 0; mi < 2; ++mi)
#pragma unroll
    for (int r = 0; r < 8; ++r) {
      const float inv = 1.0f / lst[mi][r];
      const int row = b * SEQ + q0 + 16 * mi + r + 8 * g;
#pragma unroll
      for (int t = 0; t < 4; ++t)
        z[(size_t)row * D_MODEL + head * D_HEAD + 16 * t + n] = (bf16)(accz[mi][t][r] * inv);
    }
}

// ---------------------------------------------------------------- output GEMM
// 64x64 tile per wave, same blocking as qkv_gemm.
__global__ void WAVE_KERNEL
out_gemm(const bf16* __restrict__ z, const bf16* __restrict__ woT,
         const float* __restrict__ bO, float* __restrict__ out) {
  const int row0 = blockIdx.x * 64;
  const int c0   = blockIdx.y * 64;

  v8f acc[4][4] = {};
  for (int k0 = 0; k0 < D_MODEL; k0 += 32) {
    if (k0 + 32 < D_MODEL)
      __builtin_prefetch(z + (size_t)row0 * D_MODEL + k0 + 32, 0, 3);  // WGP-scope
    v16bf a[4];
#pragma unroll
    for (int mi = 0; mi < 4; ++mi)
      a[mi] = load_a_frag(z + (size_t)(row0 + 16 * mi) * D_MODEL + k0, D_MODEL);
#pragma unroll
    for (int t = 0; t < 4; ++t) {
      v16bf b = load_bT_frag(woT + (size_t)(c0 + 16 * t) * D_MODEL + k0, D_MODEL);
#pragma unroll
      for (int mi = 0; mi < 4; ++mi)
        acc[mi][t] = wmma_bf16(a[mi], b, acc[mi][t]);
    }
  }

  const int lane = threadIdx.x & 31;
  const int n = lane & 15, g = lane >> 4;
#pragma unroll
  for (int mi = 0; mi < 4; ++mi)
#pragma unroll
    for (int t = 0; t < 4; ++t) {
      const float bias = bO[c0 + 16 * t + n];
#pragma unroll
      for (int r = 0; r < 8; ++r)
        out[(size_t)(row0 + 16 * mi + r + 8 * g) * D_MODEL + c0 + 16 * t + n] =
            acc[mi][t][r] + bias;
    }
}

// ---------------------------------------------------------------- launch

extern "C" void kernel_launch(void* const* d_in, const int* in_sizes, int n_in,
                              void* d_out, int out_size, void* d_ws, size_t ws_size,
                              hipStream_t stream) {
  const float* x   = (const float*)d_in[0];
  const float* W_Q = (const float*)d_in[1];
  const float* W_K = (const float*)d_in[2];
  const float* W_V = (const float*)d_in[3];
  const float* W_O = (const float*)d_in[4];
  const float* b_Q = (const float*)d_in[5];
  const float* b_K = (const float*)d_in[6];
  const float* b_V = (const float*)d_in[7];
  const float* b_O = (const float*)d_in[8];
  float* out = (float*)d_out;

  // workspace layout (bf16 unless noted), 256B-aligned regions, ~68 MB total
  char* ws = (char*)d_ws;
  size_t off = 0;
  auto take = [&](size_t bytes) { char* p = ws + off; off = (off + bytes + 255) & ~(size_t)255; return p; };
  bf16* xb    = (bf16*)take((size_t)ROWS * D_MODEL * 2);        // 12.6 MB
  bf16* wqkvT = (bf16*)take((size_t)NQKV * D_MODEL * 2);        //  3.5 MB
  bf16* woT   = (bf16*)take((size_t)D_MODEL * D_MODEL * 2);     //  1.2 MB
  bf16* qw    = (bf16*)take((size_t)BATCH * N_HEADS * SEQ * D_HEAD * 2);
  bf16* kw    = (bf16*)take((size_t)BATCH * N_HEADS * SEQ * D_HEAD * 2);
  bf16* vwT   = (bf16*)take((size_t)BATCH * N_HEADS * D_HEAD * SEQ * 2);
  bf16* zw    = (bf16*)take((size_t)ROWS * D_MODEL * 2);

  // 1) precision conversion + weight repack (B^T layouts)
  {
    int n = ROWS * D_MODEL;
    prep_x_bf16<<<1024, 256, 0, stream>>>(x, xb, n);
    int nw = NQKV * D_MODEL;
    pack_wqkv<<<(nw + 255) / 256, 256, 0, stream>>>(W_Q, W_K, W_V, wqkvT);
    int no = D_MODEL * D_MODEL;
    pack_wo<<<(no + 255) / 256, 256, 0, stream>>>(W_O, woT);
  }

  // 2) fused QKV projection GEMM (bf16 WMMA, fp32 accum)
  qkv_gemm<<<dim3(ROWS / 64, NQKV / 64), 32, 0, stream>>>(
      xb, wqkvT, b_Q, b_K, b_V, qw, kw, vwT);

  // 3) causal flash attention per (b, head, 32-query tile)
  attn_kernel<<<dim3(SEQ / 32, BATCH * N_HEADS), 32, 0, stream>>>(qw, kw, vwT, zw);

  // 4) output projection GEMM + bias, fp32 out
  out_gemm<<<dim3(ROWS / 64, D_MODEL / 64), 32, 0, stream>>>(zw, woT, b_O, out);
}
